// AttentionBlock_86517821214902
// MI455X (gfx1250) — matrix operand order repaired and checked by reading, not compiled, not run on hardware
//
#include <hip/hip_runtime.h>
#include <hip/hip_bf16.h>

// ---------------------------------------------------------------------------
// AttentionBlock for MI455X (gfx1250): bf16 WMMA, wave32, async-LDS staging.
// B=8, T=2048, C=512.  softmax is over the QUERY axis (axis=1) -> two-pass:
// column stats (online max/sum-exp down each masked column), then P@V with
// stats known.  ~95 GFLOP vs ~100MB HBM => compute-bound => everything WMMA.
// ---------------------------------------------------------------------------

#define B_  8
#define T_  2048
#define C_  512
#define M_  (B_ * T_)          // 16384 rows total

typedef __bf16 bf16_t;
typedef bf16_t v16bf __attribute__((ext_vector_type(16)));
typedef bf16_t v8bf  __attribute__((ext_vector_type(8)));
typedef bf16_t v4bf  __attribute__((ext_vector_type(4)));
typedef float  v8f   __attribute__((ext_vector_type(8)));
typedef float  v4f   __attribute__((ext_vector_type(4)));

#if defined(__has_builtin)
#if __has_builtin(__builtin_amdgcn_global_load_async_to_lds_b128) && \
    __has_builtin(__builtin_amdgcn_s_wait_asynccnt)
#define USE_ASYNC_COPY 1
#endif
#endif

#ifdef USE_ASYNC_COPY
// ROCm clang-22 declares the async-LDS builtin with int4-vector pointees:
//   (global i32x4* src, local i32x4* dst, imm offset, imm cpol)
typedef int i32x4 __attribute__((vector_size(16)));
typedef __attribute__((address_space(1))) i32x4* gv4i_p;
typedef __attribute__((address_space(3))) i32x4* lv4i_p;
#endif

__device__ __forceinline__ v8f wmma_bf16(v16bf a, v16bf b, v8f c) {
  // D = A(16x32) * B(32x16) + C, fp32 accumulate
  return __builtin_amdgcn_wmma_f32_16x16x32_bf16(
      false, a, false, b, (short)0, c, false, false);
}

// A-matrix 16x32 bf16 tile from row-major memory (global or LDS).
// lane<16: row=lane, K=0..7 (elems0..7), 16..23 (elems8..15); lane>=16: +8.
__device__ __forceinline__ v16bf load_a(const bf16_t* base, int stride) {
  int lane = threadIdx.x & 31;
  const bf16_t* p = base + (size_t)(lane & 15) * stride + ((lane >> 4) << 3);
  v16bf r;
  ((v8bf*)&r)[0] = *(const v8bf*)p;
  ((v8bf*)&r)[1] = *(const v8bf*)(p + 16);
  return r;
}

// B-matrix 32x16 bf16 tile where memory rows are B *columns* (B = rows^T).
// lane<16: col=lane, K=0..7 (elems0..7), 16..23 (elems8..15); lane>=16: +8.
__device__ __forceinline__ v16bf load_b(const bf16_t* base, int stride) {
  int lane = threadIdx.x & 31;
  const bf16_t* p = base + (size_t)(lane & 15) * stride + ((lane >> 4) << 3);
  v16bf r;
  ((v8bf*)&r)[0] = *(const v8bf*)p;
  ((v8bf*)&r)[1] = *(const v8bf*)(p + 16);
  return r;
}

// --------------------------- kernel 1: prep x ------------------------------
__global__ void __launch_bounds__(256) k_prep_x(const float* __restrict__ x,
                                                bf16_t* __restrict__ xb,
                                                float* __restrict__ out) {
  size_t i4 = ((size_t)blockIdx.x * blockDim.x + threadIdx.x) << 2;
  if (i4 >= (size_t)M_ * C_) return;
  v4f v = *(const v4f*)(x + i4);
  v4bf b;
  b[0] = (bf16_t)v[0]; b[1] = (bf16_t)v[1];
  b[2] = (bf16_t)v[2]; b[3] = (bf16_t)v[3];
  *(v4bf*)(xb + i4) = b;
  size_t row = i4 >> 9;
  int    c   = (int)(i4 & 511);
  *(v4f*)(out + row * 1024 + c) = v;   // concat: x occupies cols 0..511
}

// --------------------------- kernel 2: prep weights ------------------------
__global__ void __launch_bounds__(256) k_prep_w(const float* __restrict__ wq,
                                                const float* __restrict__ wk,
                                                const float* __restrict__ wv,
                                                bf16_t* __restrict__ wb) {
  const size_t WSZ = (size_t)C_ * C_;
  size_t i4 = ((size_t)blockIdx.x * blockDim.x + threadIdx.x) << 2;
  if (i4 >= 3 * WSZ) return;
  const float* src;
  size_t local;
  if (i4 < WSZ)          { src = wq; local = i4; }
  else if (i4 < 2 * WSZ) { src = wk; local = i4 - WSZ; }
  else                   { src = wv; local = i4 - 2 * WSZ; }
  v4f v = *(const v4f*)(src + local);
  v4bf b;
  b[0] = (bf16_t)v[0]; b[1] = (bf16_t)v[1];
  b[2] = (bf16_t)v[2]; b[3] = (bf16_t)v[3];
  *(v4bf*)(wb + i4) = b;
}

// --------------------------- kernel 3: QKV projection ----------------------
// One wave: 16(M) x 64(N) tile of q, k, or v.  Software-pipelined k loop.
__global__ void __launch_bounds__(256) k_qkv(const bf16_t* __restrict__ xb,
                                             const bf16_t* __restrict__ wb,
                                             const float* __restrict__ bq,
                                             const float* __restrict__ bk,
                                             const float* __restrict__ bv,
                                             bf16_t* __restrict__ qb,
                                             bf16_t* __restrict__ kb,
                                             bf16_t* __restrict__ vtb) {
  int lane = threadIdx.x & 31;
  int W = blockIdx.x * 8 + (threadIdx.x >> 5);
  int s   = W / 8192;                 // 0=q,1=k,2=v
  int rem = W - s * 8192;
  int m0 = (rem >> 3) << 4;
  int n0 = (rem & 7) << 6;
  const bf16_t* wmat = wb + (size_t)s * C_ * C_;
  const bf16_t* arow = xb + (size_t)m0 * C_;
  const float* bias = (s == 0) ? bq : ((s == 1) ? bk : bv);
  int r = lane & 15, hi = lane >> 4;

  v8f acc[4] = {v8f{}, v8f{}, v8f{}, v8f{}};
  v16bf a = load_a(arow, C_);
  v16bf bm[4];
#pragma unroll
  for (int t = 0; t < 4; ++t) bm[t] = load_b(wmat + (size_t)(n0 + 16 * t) * C_, C_);

#pragma unroll
  for (int kk = 0; kk < C_; kk += 32) {
    v16bf an = a, bn[4] = {bm[0], bm[1], bm[2], bm[3]};
    if (kk + 32 < C_) {                       // compile-time per unrolled iter
      an = load_a(arow + kk + 32, C_);
#pragma unroll
      for (int t = 0; t < 4; ++t)
        bn[t] = load_b(wmat + (size_t)(n0 + 16 * t) * C_ + kk + 32, C_);
    }
#pragma unroll
    for (int t = 0; t < 4; ++t) acc[t] = wmma_bf16(a, bm[t], acc[t]);
    a = an;
#pragma unroll
    for (int t = 0; t < 4; ++t) bm[t] = bn[t];
  }

#pragma unroll
  for (int t = 0; t < 4; ++t) {
    int col = n0 + 16 * t + r;
    float bia = bias[col];
#pragma unroll
    for (int g = 0; g < 8; ++g) {
      int row = m0 + g + 8 * hi;
      float val = acc[t][g] + bia;
      if (s == 0) {
        qb[(size_t)row * C_ + col] = (bf16_t)val;
      } else if (s == 1) {
        kb[(size_t)row * C_ + col] = (bf16_t)val;
      } else {
        int bb = row >> 11, tt = row & (T_ - 1);
        vtb[((size_t)bb * C_ + col) * T_ + tt] = (bf16_t)val;   // v^T
      }
    }
  }
}

// --------------------------- kernel 4: column softmax stats ----------------
// One wave owns 16 columns j, walks i-tiles from the diagonal down with an
// online (max, sum-exp).  z = score/sqrt(512); mask j>i before scaling.
__global__ void __launch_bounds__(256) k_colstats(const bf16_t* __restrict__ qb,
                                                  const bf16_t* __restrict__ kb,
                                                  float* __restrict__ cmax,
                                                  float* __restrict__ cinv) {
  const float scale = 0.044194173824159216f;
  int lane = threadIdx.x & 31;
  int W = blockIdx.x * 8 + (threadIdx.x >> 5);
  int b  = W >> 7;
  int j0 = (W & 127) << 4;
  const bf16_t* qrow = qb + (size_t)b * T_ * C_;
  const bf16_t* krow = kb + (size_t)b * T_ * C_;
  int r = lane & 15, hi = lane >> 4;

  float rm = -__builtin_inff(), rs = 0.0f;
  for (int i0 = j0; i0 < T_; i0 += 16) {
    v8f acc = v8f{};
    v16bf a  = load_a(qrow + (size_t)i0 * C_, C_);
    v16bf bm = load_b(krow + (size_t)j0 * C_, C_);
#pragma unroll
    for (int kk = 0; kk < C_; kk += 32) {
      v16bf an = a, bn = bm;
      if (kk + 32 < C_) {
        an = load_a(qrow + (size_t)i0 * C_ + kk + 32, C_);
        bn = load_b(krow + (size_t)j0 * C_ + kk + 32, C_);
      }
      acc = wmma_bf16(a, bm, acc);
      a = an; bm = bn;
    }
    int j = j0 + r;
    float z[8], tm = -__builtin_inff();
#pragma unroll
    for (int g = 0; g < 8; ++g) {
      int i = i0 + g + 8 * hi;
      z[g] = (i >= j) ? acc[g] * scale : -__builtin_inff();
      tm = fmaxf(tm, z[g]);
    }
    tm = fmaxf(tm, __shfl_xor(tm, 16, 32));   // columns split over lane halves
    float ts = 0.0f;
#pragma unroll
    for (int g = 0; g < 8; ++g) ts += __expf(z[g] - tm);
    ts += __shfl_xor(ts, 16, 32);
    float nm = fmaxf(rm, tm);
    rs = rs * __expf(rm - nm) + ts * __expf(tm - nm);
    rm = nm;
  }
  if (lane < 16) {
    cmax[(size_t)b * T_ + j0 + lane] = rm;
    cinv[(size_t)b * T_ + j0 + lane] = 1.0f / rs;
  }
}

// --------------------------- kernel 5: P @ V (cooperative block) -----------
// Block = (batch, 16-row i-tile), 8 waves cover all 512 n (n0 = wid*64).
// q tile (16x512, 16KB, contiguous) staged to LDS once via async-LDS DMA.
// Per 256-wide j block: wave w computes S/P for j slice [jb*256+w*32, +32)
// (no duplicated S work), P goes to shared LDS tile, barrier, then every
// wave runs P@v^T over all live 32-j sub-tiles for its n slice.
__global__ void __launch_bounds__(256) k_attn_out(const bf16_t* __restrict__ qb,
                                                  const bf16_t* __restrict__ kb,
                                                  const bf16_t* __restrict__ vtb,
                                                  const float* __restrict__ cmax,
                                                  const float* __restrict__ cinv,
                                                  float* __restrict__ out) {
  __shared__ __align__(16) bf16_t qlds[16 * 512];   // 16 KB: q i-tile
  __shared__ __align__(16) bf16_t plds[16 * 256];   //  8 KB: P block tile
  const float scale = 0.044194173824159216f;
  int lane = threadIdx.x & 31;
  int wid  = threadIdx.x >> 5;
  int b  = (int)blockIdx.x >> 7;          // 1024 blocks = 8 batches * 128 tiles
  int i0 = ((int)blockIdx.x & 127) << 4;
  int n0 = wid << 6;
  const bf16_t* qrow = qb  + (size_t)b * T_ * C_;
  const bf16_t* krow = kb  + (size_t)b * T_ * C_;
  const bf16_t* vt   = vtb + (size_t)b * C_ * T_;
  const float*  cm   = cmax + (size_t)b * T_;
  const float*  ci   = cinv + (size_t)b * T_;
  int r = lane & 15, hi = lane >> 4;
  int tid = threadIdx.x;

  // ---- stage q tile (rows i0..i0+15 are contiguous: 8192 bf16) into LDS ----
  const bf16_t* qsrc = qrow + (size_t)i0 * C_;
#ifdef USE_ASYNC_COPY
#pragma unroll
  for (int it = 0; it < 4; ++it) {
    size_t eoff = ((size_t)(tid + (it << 8))) << 3;   // 16B chunks
    __builtin_amdgcn_global_load_async_to_lds_b128(
        (gv4i_p)(uintptr_t)(qsrc + eoff),
        (lv4i_p)(uint32_t)(uintptr_t)(qlds + eoff),
        0, 0);
  }
  __builtin_amdgcn_s_wait_asynccnt(0);
#else
#pragma unroll
  for (int it = 0; it < 4; ++it) {
    size_t eoff = ((size_t)(tid + (it << 8))) << 3;
    *(v8bf*)(qlds + eoff) = *(const v8bf*)(qsrc + eoff);
  }
#endif
  __syncthreads();

  v8f acc[4] = {v8f{}, v8f{}, v8f{}, v8f{}};
  int njb = (i0 >> 8) + 1;                 // 256-wide j blocks up to diagonal
  for (int jb = 0; jb < njb; ++jb) {
    int jbase = jb << 8;
    int j0w = jbase + (wid << 5);          // this wave's 32-wide j slice
    bool active = (j0w <= i0 + 15);        // uniform per wave

    v8f sacc[2] = {v8f{}, v8f{}};
    if (active) {
      v16bf a  = load_a(qlds, C_);
      v16bf b0 = load_b(krow + (size_t)j0w * C_, C_);
      v16bf b1 = load_b(krow + (size_t)(j0w + 16) * C_, C_);
#pragma unroll
      for (int kk = 0; kk < C_; kk += 32) {
        v16bf an = a, bn0 = b0, bn1 = b1;
        if (kk + 32 < C_) {
          an  = load_a(qlds + kk + 32, C_);
          bn0 = load_b(krow + (size_t)j0w * C_ + kk + 32, C_);
          bn1 = load_b(krow + (size_t)(j0w + 16) * C_ + kk + 32, C_);
        }
        sacc[0] = wmma_bf16(a, b0, sacc[0]);
        sacc[1] = wmma_bf16(a, b1, sacc[1]);
        a = an; b0 = bn0; b1 = bn1;
      }
    }
    // P = exp(z - colmax)*colsum^-1 for i>=j else 0  (D-layout -> LDS tile)
#pragma unroll
    for (int jj = 0; jj < 2; ++jj) {
      int j = j0w + jj * 16 + r;
      float m_ = cm[j], v_ = ci[j];
#pragma unroll
      for (int g = 0; g < 8; ++g) {
        int i = i0 + g + 8 * hi;
        float p = (i >= j) ? __expf(sacc[jj][g] * scale - m_) * v_ : 0.0f;
        plds[(g + 8 * hi) * 256 + (wid << 5) + jj * 16 + r] = (bf16_t)p;
      }
    }
    __syncthreads();

    // consume live P sub-tiles against this wave's 64-n slice of v^T
    int njj = ((i0 + 15 - jbase) >> 5) + 1;   // uniform across block
    if (njj > 8) njj = 8;
    for (int jj8 = 0; jj8 < njj; ++jj8) {
      v16bf pa = load_a(plds + (jj8 << 5), 256);
#pragma unroll
      for (int t = 0; t < 4; ++t) {
        v16bf bm = load_b(vt + (size_t)(n0 + 16 * t) * T_ + jbase + (jj8 << 5), T_);
        acc[t] = wmma_bf16(pa, bm, acc[t]);
      }
    }
    __syncthreads();                        // protect plds before next jb
  }

#pragma unroll
  for (int t = 0; t < 4; ++t) {
    int col = 512 + n0 + 16 * t + r;
#pragma unroll
    for (int g = 0; g < 8; ++g) {
      size_t row = (size_t)b * T_ + i0 + g + 8 * hi;
      out[row * 1024 + col] = acc[t][g];
    }
  }
}

// --------------------------- launch ---------------------------------------
extern "C" void kernel_launch(void* const* d_in, const int* in_sizes, int n_in,
                              void* d_out, int out_size, void* d_ws, size_t ws_size,
                              hipStream_t stream) {
  const float* x  = (const float*)d_in[0];
  const float* Wq = (const float*)d_in[1];
  const float* bq = (const float*)d_in[2];
  const float* Wk = (const float*)d_in[3];
  const float* bk = (const float*)d_in[4];
  const float* Wv = (const float*)d_in[5];
  const float* bv = (const float*)d_in[6];
  float* out = (float*)d_out;

  char* ws = (char*)d_ws;
  bf16_t* xb  = (bf16_t*)ws;                                   // 16 MB
  bf16_t* wb  = (bf16_t*)(ws + (size_t)16777216);              // 1.5 MB
  bf16_t* qb  = (bf16_t*)(ws + (size_t)16777216 + 1572864);    // 16 MB
  bf16_t* kb  = qb + (size_t)M_ * C_;                          // 16 MB
  bf16_t* vtb = kb + (size_t)M_ * C_;                          // 16 MB
  float*  cmaxp = (float*)(vtb + (size_t)M_ * C_);             // 64 KB
  float*  cinvp = cmaxp + (size_t)B_ * T_;                     // 64 KB

  k_prep_x<<<8192, 256, 0, stream>>>(x, xb, out);
  k_prep_w<<<768, 256, 0, stream>>>(Wq, Wk, Wv, wb);
  k_qkv<<<3072, 256, 0, stream>>>(xb, wb, bq, bk, bv, qb, kb, vtb);
  k_colstats<<<128, 256, 0, stream>>>(qb, kb, cmaxp, cinvp);
  k_attn_out<<<1024, 256, 0, stream>>>(qb, kb, vtb, cmaxp, cinvp, out);
}
